// GraphSAGEModel_11793980195325
// MI455X (gfx1250) — matrix operand order repaired and checked by reading, not compiled, not run on hardware
//
#include <hip/hip_runtime.h>
#include <hip/hip_bf16.h>

// ---------------------------------------------------------------------------
// GraphSAGE link-prediction forward for MI455X (gfx1250, wave32, WMMA)
//   - f16 WMMA (v_wmma_f32_16x16x32_f16), fp32 accumulate
//   - weights pre-swizzled to B-fragment layout -> ds_load_b128 operand feeds
//   - sched_group_barrier pins DS-read / WMMA pipelining per K-chunk
// ---------------------------------------------------------------------------

typedef __attribute__((ext_vector_type(16))) _Float16 v16h;
typedef __attribute__((ext_vector_type(8)))  _Float16 v8h;
typedef __attribute__((ext_vector_type(4)))  _Float16 v4h;
typedef __attribute__((ext_vector_type(8)))  float    v8f;
typedef __attribute__((ext_vector_type(4)))  float    v4f;

#define HD 128          // feature width (D == H == 128)
#define SAGE_ROWS 64    // rows per block in GEMM kernels
#define DEC_EDGES 64    // edges per block in decoder
#define WELEMS (HD * HD)

// ---- WMMA helper ----------------------------------------------------------
__device__ __forceinline__ v8f wmma_f16(v16h a, v16h b, v8f c) {
    // D = A(16x32 f16) * B(32x16 f16) + C(16x16 f32)
    return __builtin_amdgcn_wmma_f32_16x16x32_f16(
        /*neg_a=*/false, a, /*neg_b=*/false, b,
        /*c_mod=*/(short)0, c, /*reuse_a=*/false, /*reuse_b=*/false);
}

__device__ __forceinline__ v16h cat8(v8h lo, v8h hi) {
    return __builtin_shufflevector(lo, hi, 0, 1, 2, 3, 4, 5, 6, 7,
                                           8, 9, 10, 11, 12, 13, 14, 15);
}

// A fragment: 16x32 f16 tile from row-major LDS [rows x 128].
// ISA 7.12.2: lane L(0..15) -> M=L,   K halves {k0..k0+7, k0+16..k0+23}, k0=kBase
//             lane L(16..31)-> M=L-16, k0=kBase+8
// Both 8-half runs are contiguous & 16B aligned -> two ds_load_b128.
__device__ __forceinline__ v16h load_a_frag(const _Float16* s, int rowBase,
                                            int kBase, int lane) {
    const int m  = rowBase + (lane & 15);
    const int k0 = kBase + ((lane < 16) ? 0 : 8);
    const v8h* p = (const v8h*)(s + m * HD + k0);
    return cat8(p[0], p[2]);       // halves k0..k0+7 and k0+16..k0+23
}

// B fragment from pre-swizzled LDS weight: each lane's 16 halves contiguous.
// swizzled index: ((kc*8 + ct)*32 + lane)*16 + h
__device__ __forceinline__ v16h load_b_frag(const _Float16* sWf, int kc,
                                            int ct, int lane) {
    const v8h* p = (const v8h*)(sWf + ((((kc << 3) + ct) << 5) + lane) * 16);
    return cat8(p[0], p[1]);
}

// one K-chunk of a 16x128 output strip: batch-load 2 A + 16 B ds_load_b128,
// then 8 back-to-back WMMAs. sched_group_barrier enforces the pipeline shape
// (DS-read group first, WMMA group after) so waits are staged, not dscnt==0.
__device__ __forceinline__ void gemm_kchunk(const _Float16* sA,
                                            const _Float16* sW, int wr,
                                            int kc, int lane, v8f acc[8]) {
    v16h a = load_a_frag(sA, wr, kc * 32, lane);
    v16h bf[8];
#pragma unroll
    for (int ct = 0; ct < 8; ++ct) bf[ct] = load_b_frag(sW, kc, ct, lane);
#pragma unroll
    for (int ct = 0; ct < 8; ++ct) acc[ct] = wmma_f16(a, bf[ct], acc[ct]);
#if __has_builtin(__builtin_amdgcn_sched_group_barrier)
    __builtin_amdgcn_sched_group_barrier(0x100, 18, 0);  // 18 DS reads
    __builtin_amdgcn_sched_group_barrier(0x008,  8, 0);  // 8 WMMA (MFMA mask)
#endif
}

// ---- weight cast + swizzle to fragment-major layout -----------------------
// dst[i], i = (((kc*8+ct)*32 + lane)*16 + h)  <-  W[k*128+n],
//   k = kc*32 + ((lane&16)>>1) + (h&7) + ((h&8)<<1), n = ct*16 + (lane&15)
__global__ void k_cast_swz(const float* __restrict__ src,
                           _Float16* __restrict__ dst) {
    int i = blockIdx.x * blockDim.x + threadIdx.x;
    if (i < WELEMS) {
        int h    = i & 15;
        int lane = (i >> 4) & 31;
        int ct   = (i >> 9) & 7;
        int kc   = i >> 12;
        int k = kc * 32 + ((lane & 16) >> 1) + (h & 7) + ((h & 8) << 1);
        int n = ct * 16 + (lane & 15);
        dst[i] = (_Float16)src[k * HD + n];
    }
}

// ---- small utility kernels ------------------------------------------------
__global__ void k_degree(const int* __restrict__ dst_idx,
                         float* __restrict__ deg, int e) {
    int i = blockIdx.x * blockDim.x + threadIdx.x;
    if (i < e) atomicAdd(&deg[dst_idx[i]], 1.0f);
}

__global__ void k_rdeg(float* __restrict__ deg, int n) {
    int i = blockIdx.x * blockDim.x + threadIdx.x;
    if (i < n) deg[i] = 1.0f / fmaxf(deg[i], 1.0f);
}

// scatter-add: agg[dst[e], f] += feats[src[e], f]   (atomics resolve in L2)
__global__ void k_scatter(const float* __restrict__ feats,
                          const int* __restrict__ src_idx,
                          const int* __restrict__ dst_idx,
                          float* __restrict__ agg, int e) {
    int i = blockIdx.x * blockDim.x + threadIdx.x;
    int edge = i >> 7;
    int f    = i & 127;
    if (edge < e) {
        int s = src_idx[edge];
        int d = dst_idx[edge];
        atomicAdd(&agg[(size_t)d * HD + f], feats[(size_t)s * HD + f]);
    }
}

// ---- SAGE layer: out = (relu?)( hself @ Wself + (agg*rdeg) @ Wneigh + b ) -
__global__ __launch_bounds__(128)
void k_sage(const float* __restrict__ hself, const float* __restrict__ agg,
            const float* __restrict__ rdeg,
            const _Float16* __restrict__ Wself,    // swizzled
            const _Float16* __restrict__ Wneigh,   // swizzled
            const float* __restrict__ bias,
            float* __restrict__ out, int n, int do_relu) {
    __shared__ __align__(16) _Float16 sW[WELEMS];          // 32 KB, swizzled
    __shared__ __align__(16) _Float16 sA[SAGE_ROWS * HD];  // 16 KB, row-major

    const int t    = threadIdx.x;
    const int lane = t & 31;
    const int wv   = t >> 5;                    // 4 waves
    const int rowBase = blockIdx.x * SAGE_ROWS;
    const int wr = wv * 16;                     // wave's local row base

    v8f acc[8];
#pragma unroll
    for (int ct = 0; ct < 8; ++ct) { v8f z = {}; acc[ct] = z; }

#pragma unroll 1
    for (int pass = 0; pass < 2; ++pass) {
        // contiguous copy of swizzled weights: b128 in / b128 out
        const uint4* wg = (const uint4*)(pass ? Wneigh : Wself);
        uint4* ws4 = (uint4*)sW;
#pragma unroll
        for (int i = t; i < WELEMS / 8; i += 128) ws4[i] = wg[i];

        // stage activations, float4 -> 4 halves
        v4h* sa4 = (v4h*)sA;
        for (int i = t; i < SAGE_ROWS * HD / 4; i += 128) {
            int r  = i >> 5;           // 32 float4 per row
            int f4 = i & 31;
            int gr = rowBase + r;
            v4f v = {};
            if (gr < n) {
                const v4f* srcp = (const v4f*)((pass ? agg : hself) +
                                               (size_t)gr * HD);
                v = srcp[f4];
                if (pass) v *= rdeg[gr];
            }
            v4h hv;
#pragma unroll
            for (int j = 0; j < 4; ++j) hv[j] = (_Float16)v[j];
            sa4[i] = hv;
        }
        __syncthreads();

#pragma unroll
        for (int kc = 0; kc < 4; ++kc)
            gemm_kchunk(sA, sW, wr, kc, lane, acc);
        __syncthreads();    // before next pass overwrites sW/sA
    }

    // epilogue: C layout: VGPR j, lane<16 -> M=j ; lane>=16 -> M=j+8 ; N=lane&15
    const int mo = (lane < 16) ? 0 : 8;
    const int nn = lane & 15;
#pragma unroll
    for (int ct = 0; ct < 8; ++ct) {
        int ncol = ct * 16 + nn;
        float bb = bias[ncol];
#pragma unroll
        for (int j = 0; j < 8; ++j) {
            int m = rowBase + wr + j + mo;
            if (m < n) {
                float v = acc[ct][j] + bb;
                if (do_relu) v = fmaxf(v, 0.0f);
                out[(size_t)m * HD + ncol] = v;
            }
        }
    }
}

// ---- edge decoder: out[e] = W3 . relu( relu((h[s]*h[d])W1+b1) W2 + b2 ) + b3
__global__ __launch_bounds__(128)
void k_decoder(const float* __restrict__ h,
               const int* __restrict__ es, const int* __restrict__ ed,
               int count,
               const _Float16* __restrict__ W1g, const float* __restrict__ b1,
               const _Float16* __restrict__ W2g, const float* __restrict__ b2,
               const float* __restrict__ W3g, const float* __restrict__ b3,
               float* __restrict__ out) {
    __shared__ __align__(16) _Float16 sW[WELEMS];          // W1 then W2
    __shared__ __align__(16) _Float16 sZ[DEC_EDGES * HD];  // z then z1
    __shared__ float sW3[HD];

    const int t    = threadIdx.x;
    const int lane = t & 31;
    const int wv   = t >> 5;
    const int eBase = blockIdx.x * DEC_EDGES;
    const int wr = wv * 16;
    const int mo = (lane < 16) ? 0 : 8;
    const int nn = lane & 15;

    if (t < HD) sW3[t] = W3g[t];
    {
        const uint4* wg = (const uint4*)W1g;
        uint4* ws4 = (uint4*)sW;
#pragma unroll
        for (int i = t; i < WELEMS / 8; i += 128) ws4[i] = wg[i];
    }
    // stage z = h[s] * h[d], float4 loads
    {
        v4h* sz4 = (v4h*)sZ;
        for (int i = t; i < DEC_EDGES * HD / 4; i += 128) {
            int e  = i >> 5;          // 32 float4 per edge
            int f4 = i & 31;
            int ge = eBase + e;
            v4f v = {};
            if (ge < count) {
                int s = es[ge], d = ed[ge];
                v4f a = ((const v4f*)(h + (size_t)s * HD))[f4];
                v4f bvec = ((const v4f*)(h + (size_t)d * HD))[f4];
                v = a * bvec;
            }
            v4h hv;
#pragma unroll
            for (int j = 0; j < 4; ++j) hv[j] = (_Float16)v[j];
            sz4[i] = hv;
        }
    }
    __syncthreads();

    v8f acc[8];
#pragma unroll
    for (int ct = 0; ct < 8; ++ct) { v8f z = {}; acc[ct] = z; }

    // GEMM1: z @ W1
#pragma unroll
    for (int kc = 0; kc < 4; ++kc)
        gemm_kchunk(sZ, sW, wr, kc, lane, acc);
    __syncthreads();    // all reads of sZ / sW complete

    // z1 = relu(acc + b1) -> sZ ; load W2 -> sW
#pragma unroll
    for (int ct = 0; ct < 8; ++ct) {
        int ncol = ct * 16 + nn;
        float bb = b1[ncol];
#pragma unroll
        for (int j = 0; j < 8; ++j) {
            sZ[(wr + j + mo) * HD + ncol] =
                (_Float16)fmaxf(acc[ct][j] + bb, 0.0f);
        }
        v8f z = {}; acc[ct] = z;
    }
    {
        const uint4* wg = (const uint4*)W2g;
        uint4* ws4 = (uint4*)sW;
#pragma unroll
        for (int i = t; i < WELEMS / 8; i += 128) ws4[i] = wg[i];
    }
    __syncthreads();

    // GEMM2: z1 @ W2
#pragma unroll
    for (int kc = 0; kc < 4; ++kc)
        gemm_kchunk(sZ, sW, wr, kc, lane, acc);

    // final: out[e] = sum_n relu(z2[e,n]+b2[n]) * W3[n] + b3
    const float b3v = b3[0];
#pragma unroll
    for (int j = 0; j < 8; ++j) {
        float p = 0.0f;
#pragma unroll
        for (int ct = 0; ct < 8; ++ct) {
            int ncol = ct * 16 + nn;
            float z2 = fmaxf(acc[ct][j] + b2[ncol], 0.0f);
            p += z2 * sW3[ncol];
        }
        // reduce across the 16-lane halves of the wave32
#pragma unroll
        for (int m = 8; m >= 1; m >>= 1) p += __shfl_xor(p, m, 16);
        int ge = eBase + wr + j + mo;
        if (nn == 0 && ge < count) out[ge] = p + b3v;
    }
}

// ---------------------------------------------------------------------------
extern "C" void kernel_launch(void* const* d_in, const int* in_sizes, int n_in,
                              void* d_out, int out_size, void* d_ws, size_t ws_size,
                              hipStream_t stream) {
    // inputs (setup_inputs order)
    const float* x        = (const float*)d_in[0];
    const int*   src      = (const int*)d_in[1];
    const int*   dst      = (const int*)d_in[2];
    const int*   pos_src  = (const int*)d_in[3];
    const int*   pos_dst  = (const int*)d_in[4];
    const int*   neg_src  = (const int*)d_in[5];
    const int*   neg_dst  = (const int*)d_in[6];
    const float* W_self0  = (const float*)d_in[7];
    const float* W_neigh0 = (const float*)d_in[8];
    const float* b0       = (const float*)d_in[9];
    const float* W_self1  = (const float*)d_in[10];
    const float* W_neigh1 = (const float*)d_in[11];
    const float* b1       = (const float*)d_in[12];
    const float* dW1      = (const float*)d_in[13];
    const float* db1      = (const float*)d_in[14];
    const float* dW2      = (const float*)d_in[15];
    const float* db2      = (const float*)d_in[16];
    const float* dW3      = (const float*)d_in[17];
    const float* db3      = (const float*)d_in[18];

    const int N  = in_sizes[0] / HD;
    const int E  = in_sizes[1];
    const int EP = in_sizes[3];
    const int EN = in_sizes[5];

    // workspace carve-up
    char*  ws = (char*)d_ws;
    size_t o  = 0;
    auto carve = [&](size_t bytes) -> void* {
        void* p = ws + o;
        o = (o + bytes + 255) & ~(size_t)255;
        return p;
    };
    const size_t WB = (size_t)WELEMS * sizeof(_Float16);
    _Float16* hWs0 = (_Float16*)carve(WB);
    _Float16* hWn0 = (_Float16*)carve(WB);
    _Float16* hWs1 = (_Float16*)carve(WB);
    _Float16* hWn1 = (_Float16*)carve(WB);
    _Float16* hW1  = (_Float16*)carve(WB);
    _Float16* hW2  = (_Float16*)carve(WB);
    float* rdeg = (float*)carve((size_t)N * sizeof(float));
    float* agg  = (float*)carve((size_t)N * HD * sizeof(float));
    float* h0   = (float*)carve((size_t)N * HD * sizeof(float));
    float* h1   = (float*)carve((size_t)N * HD * sizeof(float));

    float* out = (float*)d_out;   // [EP | EN]

    const int TB = 256;
    const int WG = (WELEMS + TB - 1) / TB;

    // 1) cast + swizzle weights to WMMA B-fragment layout (f16)
    k_cast_swz<<<WG, TB, 0, stream>>>(W_self0,  hWs0);
    k_cast_swz<<<WG, TB, 0, stream>>>(W_neigh0, hWn0);
    k_cast_swz<<<WG, TB, 0, stream>>>(W_self1,  hWs1);
    k_cast_swz<<<WG, TB, 0, stream>>>(W_neigh1, hWn1);
    k_cast_swz<<<WG, TB, 0, stream>>>(dW1,      hW1);
    k_cast_swz<<<WG, TB, 0, stream>>>(dW2,      hW2);

    // 2) degree -> reciprocal
    hipMemsetAsync(rdeg, 0, (size_t)N * sizeof(float), stream);
    k_degree<<<(E + TB - 1) / TB, TB, 0, stream>>>(dst, rdeg, E);
    k_rdeg<<<(N + TB - 1) / TB, TB, 0, stream>>>(rdeg, N);

    // 3) layer 0: aggregate x, then GEMM + relu
    hipMemsetAsync(agg, 0, (size_t)N * HD * sizeof(float), stream);
    {
        long long tot = (long long)E * HD;
        k_scatter<<<(int)((tot + TB - 1) / TB), TB, 0, stream>>>(x, src, dst, agg, E);
    }
    k_sage<<<(N + SAGE_ROWS - 1) / SAGE_ROWS, 128, 0, stream>>>(
        x, agg, rdeg, hWs0, hWn0, b0, h0, N, /*relu=*/1);

    // 4) layer 1: aggregate h0, GEMM (no relu)
    hipMemsetAsync(agg, 0, (size_t)N * HD * sizeof(float), stream);
    {
        long long tot = (long long)E * HD;
        k_scatter<<<(int)((tot + TB - 1) / TB), TB, 0, stream>>>(h0, src, dst, agg, E);
    }
    k_sage<<<(N + SAGE_ROWS - 1) / SAGE_ROWS, 128, 0, stream>>>(
        h0, agg, rdeg, hWs1, hWn1, b1, h1, N, /*relu=*/0);

    // 5) decoders
    k_decoder<<<(EP + DEC_EDGES - 1) / DEC_EDGES, 128, 0, stream>>>(
        h1, pos_src, pos_dst, EP, hW1, db1, hW2, db2, dW3, db3, out);
    k_decoder<<<(EN + DEC_EDGES - 1) / DEC_EDGES, 128, 0, stream>>>(
        h1, neg_src, neg_dst, EN, hW1, db1, hW2, db2, dW3, db3, out + EP);
}
